// MyGCN_38379827757077
// MI455X (gfx1250) — compile-verified, matrix operand order checked by reading, not run on hardware
//
#include <hip/hip_runtime.h>
#include <hip/hip_bf16.h>

typedef __attribute__((ext_vector_type(16))) _Float16 v16h;
typedef __attribute__((ext_vector_type(8)))  _Float16 v8h;
typedef __attribute__((ext_vector_type(4)))  _Float16 v4h;
typedef __attribute__((ext_vector_type(8)))  float    v8f;
typedef __attribute__((ext_vector_type(4)))  float    v4f;

#define B_GRAPHS 256
#define N_ROI    200
#define N_NODES  (B_GRAPHS * N_ROI)         // 51200
#define EDGE_DEG 32
#define N_EDGES  ((long)N_NODES * EDGE_DEG) // 1,638,400

// =====================================================================
// WMMA GEMM: C[M,N] = act(A[M,K] @ W[K,N] + bias)
// Block = 128 threads (4 waves) -> one 16(M) x 64(N) tile.
// A tile 16x32 f16 in LDS (shared by 4 waves); B tile staged transposed
// [col][k] so each lane's fragment is two contiguous 8-half runs
// (two ds_load_b128 per matrix). Unguarded main K-loop + one guarded
// tail step; guarded loads use clamp+mask-multiply so they stay
// unconditional (no exec-masked serialized loads).
// Requires: M % 16 == 0, K % 4 == 0 (true at all call sites).
// ACT: 0=none 1=relu 2=leaky_relu(0.01) 3=sigmoid
// =====================================================================
template<bool GUARD>
__device__ __forceinline__ v8f mma_step(
    v8f acc, int k0, int K, int N, float nmask,
    const float* __restrict__ aRow, int aOff,
    const float* __restrict__ wCol, int bOff,
    _Float16* __restrict__ sAst, _Float16* __restrict__ sBst,
    const _Float16* __restrict__ pa, const _Float16* __restrict__ pb)
{
  // ---- stage A: this thread's fixed (row, k-quad), one b128 load ----
  float a0, a1, a2, a3;
  if (!GUARD) {
    const v4f av = *(const v4f*)(aRow + k0 + aOff);
    a0 = av[0]; a1 = av[1]; a2 = av[2]; a3 = av[3];
  } else {
    const int kb = k0 + aOff;
    int k;
    k = kb + 0; a0 = aRow[k < K ? k : K - 1] * ((k < K) ? 1.0f : 0.0f);
    k = kb + 1; a1 = aRow[k < K ? k : K - 1] * ((k < K) ? 1.0f : 0.0f);
    k = kb + 2; a2 = aRow[k < K ? k : K - 1] * ((k < K) ? 1.0f : 0.0f);
    k = kb + 3; a3 = aRow[k < K ? k : K - 1] * ((k < K) ? 1.0f : 0.0f);
  }
  {
    v4h ah = { (_Float16)a0, (_Float16)a1, (_Float16)a2, (_Float16)a3 };
    *(v4h*)sAst = ah;                       // ds_store_b64
  }
  // ---- stage B: fixed col, 16 k values in 4 quads, coalesced loads ----
#pragma unroll
  for (int i = 0; i < 4; ++i) {
    float w0, w1, w2, w3;
    const int kb = k0 + bOff + i * 8;
    if (!GUARD) {
      w0 = wCol[(kb + 0) * N] * nmask;
      w1 = wCol[(kb + 1) * N] * nmask;
      w2 = wCol[(kb + 2) * N] * nmask;
      w3 = wCol[(kb + 3) * N] * nmask;
    } else {
      int k;
      k = kb + 0; w0 = wCol[(k < K ? k : K - 1) * N] * ((k < K) ? nmask : 0.0f);
      k = kb + 1; w1 = wCol[(k < K ? k : K - 1) * N] * ((k < K) ? nmask : 0.0f);
      k = kb + 2; w2 = wCol[(k < K ? k : K - 1) * N] * ((k < K) ? nmask : 0.0f);
      k = kb + 3; w3 = wCol[(k < K ? k : K - 1) * N] * ((k < K) ? nmask : 0.0f);
    }
    v4h bh = { (_Float16)w0, (_Float16)w1, (_Float16)w2, (_Float16)w3 };
    *(v4h*)(sBst + i * 8) = bh;             // ds_store_b64
  }
  __syncthreads();

  // ---- fragments: two ds_load_b128 per matrix per lane ----
  const v8h a_lo = *(const v8h*)(pa);
  const v8h a_hi = *(const v8h*)(pa + 16);
  const v8h b_lo = *(const v8h*)(pb);
  const v8h b_hi = *(const v8h*)(pb + 16);
  v16h a, b;
#pragma unroll
  for (int i = 0; i < 8; ++i) {
    a[i] = a_lo[i]; a[i + 8] = a_hi[i];
    b[i] = b_lo[i]; b[i + 8] = b_hi[i];
  }
  // 8 args: (neg_a, A, neg_b, B, c_mod, C, reuse_a, reuse_b)
  acc = __builtin_amdgcn_wmma_f32_16x16x32_f16(false, a, false, b,
                                               (short)0, acc, false, false);
  __syncthreads();
  return acc;
}

template<int ACT>
__global__ void __launch_bounds__(128)
gemm_wmma_f16(const float* __restrict__ A, const float* __restrict__ W,
              const float* __restrict__ bias, float* __restrict__ C,
              int M, int N, int K)
{
  __shared__ __align__(16) _Float16 lA[16 * 32];   // [row][k]
  __shared__ __align__(16) _Float16 lB[64 * 32];   // [col][k] (transposed)

  const int t     = threadIdx.x;
  const int wave  = t >> 5;
  const int lane  = t & 31;
  const int hlf   = lane >> 4;      // 0: K in {0..7,16..23}; 1: {8..15,24..31}
  const int idx16 = lane & 15;
  const int tileM = blockIdx.y * 16;
  const int tileN = blockIdx.x * 64;

  // thread-fixed staging geometry
  const int aRowIdx = t >> 3;            // 0..15
  const int aOff    = (t & 7) << 2;      // k-quad within 32
  const float* aRow = A + (size_t)(tileM + aRowIdx) * K;

  const int bCol    = t & 63;
  const int bOff    = (t >> 6) << 2;     // 0 or 4
  const int nn      = tileN + bCol;
  const int ncl     = (nn < N) ? nn : (N - 1);
  const float nmask = (nn < N) ? 1.0f : 0.0f;
  const float* wCol = W + ncl;           // element k lives at wCol[k*N]

  _Float16* sAst = &lA[aRowIdx * 32 + aOff];
  _Float16* sBst = &lB[bCol * 32 + bOff];
  const _Float16* pa = &lA[idx16 * 32 + hlf * 8];
  const _Float16* pb = &lB[(wave * 16 + idx16) * 32 + hlf * 8];

  v8f acc = {};
  int k0 = 0;
  for (; k0 + 32 <= K; k0 += 32) {
    if (k0 + 32 < K)  // uniform: prefetch next A K-slice (global_prefetch_b8)
      __builtin_prefetch(aRow + k0 + 32, 0, 1);
    acc = mma_step<false>(acc, k0, K, N, nmask, aRow, aOff, wCol, bOff,
                          sAst, sBst, pa, pb);
  }
  if (k0 < K)         // K % 32 != 0 tail, runs at most once
    acc = mma_step<true>(acc, k0, K, N, nmask, aRow, aOff, wCol, bOff,
                         sAst, sBst, pa, pb);

  const int n = tileN + wave * 16 + idx16;
  if (n < N) {
    const float bv = bias ? bias[n] : 0.0f;
#pragma unroll
    for (int r = 0; r < 8; ++r) {
      const int mm = tileM + r + hlf * 8;   // ISA 32-bit C/D 16x16 layout
      float v = acc[r] + bv;
      if      (ACT == 1) v = fmaxf(v, 0.0f);
      else if (ACT == 2) v = (v > 0.0f) ? v : 0.01f * v;
      else if (ACT == 3) v = 1.0f / (1.0f + __expf(-v));
      C[(long)mm * N + n] = v;
    }
  }
}

// ======================= GCN support kernels =========================
__global__ void fill_k(float* __restrict__ p, float v, long n) {
  long i = (long)blockIdx.x * blockDim.x + threadIdx.x;
  if (i < n) p[i] = v;
}

__global__ void deg_accum_k(const int* __restrict__ dst, const float* __restrict__ ew,
                            float* __restrict__ deg, long E) {
  long e = (long)blockIdx.x * blockDim.x + threadIdx.x;
  if (e < E) atomicAdd(&deg[dst[e]], ew[e]);
}

__global__ void deg_inv_k(float* __restrict__ deg, int n) {
  int i = blockIdx.x * blockDim.x + threadIdx.x;
  if (i < n) {
    float d = deg[i];
    deg[i] = (d > 0.0f) ? rsqrtf(fmaxf(d, 1e-12f)) : 0.0f;
  }
}

__global__ void rowbias_init_k(float* __restrict__ out, const float* __restrict__ b,
                               int rows, int F) {
  long i = (long)blockIdx.x * blockDim.x + threadIdx.x;
  if (i < (long)rows * F) out[i] = b[(int)(i % F)];
}

// out[dst] += hw[src] * dinv[src]*ew*dinv[dst], threads laid out (edge, feature)
__global__ void gcn_scatter_k(const float* __restrict__ hw, const int* __restrict__ src,
                              const int* __restrict__ dst, const float* __restrict__ ew,
                              const float* __restrict__ dinv, float* __restrict__ out,
                              long E, int F) {
  long t = (long)blockIdx.x * blockDim.x + threadIdx.x;
  if (t >= E * (long)F) return;
  long e = t / F;
  int  f = (int)(t - e * F);
  int  s = src[e], d = dst[e];
  float norm = dinv[s] * ew[e] * dinv[d];
  atomicAdd(&out[(long)d * F + f], hw[(long)s * F + f] * norm);
}

// self loop: out[i,f] += hw[i,f] * dinv[i]^2   (weight 1)
__global__ void selfloop_k(const float* __restrict__ hw, const float* __restrict__ dinv,
                           float* __restrict__ out, int rows, int F) {
  long t = (long)blockIdx.x * blockDim.x + threadIdx.x;
  if (t >= (long)rows * F) return;
  int i = (int)(t / F);
  float dv = dinv[i];
  out[t] += hw[t] * dv * dv;
}

__global__ void relu_k(float* __restrict__ x, long n) {
  long i = (long)blockIdx.x * blockDim.x + threadIdx.x;
  if (i < n) x[i] = fmaxf(x[i], 0.0f);
}

// per-graph [max | mean] pooling into feat[:, off .. off+2F)
__global__ void pool_k(const float* __restrict__ h, float* __restrict__ feat,
                       int F, int stride, int off) {
  int t = blockIdx.x * blockDim.x + threadIdx.x;
  if (t >= B_GRAPHS * F) return;
  int g = t / F, f = t - g * F;
  const float* p = h + (long)g * N_ROI * F + f;
  float mx = -3.4e38f, sum = 0.0f;
  for (int r = 0; r < N_ROI; ++r) {
    float v = p[(long)r * F];
    mx = fmaxf(mx, v);
    sum += v;
  }
  feat[(long)g * stride + off + f]     = mx;
  feat[(long)g * stride + off + F + f] = sum * (1.0f / N_ROI);
}

// ======================= BN / elementwise ============================
__global__ void bn_stats_k(const float* __restrict__ x, float* __restrict__ mean,
                           float* __restrict__ var, int rows, int cols) {
  int c = blockIdx.x * blockDim.x + threadIdx.x;
  if (c >= cols) return;
  float mu = 0.0f;
  for (int r = 0; r < rows; ++r) mu += x[(long)r * cols + c];
  mu /= rows;
  float v = 0.0f;
  for (int r = 0; r < rows; ++r) {
    float d = x[(long)r * cols + c] - mu;
    v += d * d;
  }
  mean[c] = mu;
  var[c]  = v / rows;
}

template<int ACT> // 0=none, 2=leaky
__global__ void bn_apply_k(const float* __restrict__ x, float* __restrict__ y,
                           const float* __restrict__ mean, const float* __restrict__ var,
                           const float* __restrict__ g, const float* __restrict__ b,
                           int rows, int cols) {
  long t = (long)blockIdx.x * blockDim.x + threadIdx.x;
  if (t >= (long)rows * cols) return;
  int c = (int)(t % cols);
  float v = (x[t] - mean[c]) * rsqrtf(var[c] + 1e-5f) * g[c] + b[c];
  if (ACT == 2) v = (v > 0.0f) ? v : 0.01f * v;
  y[t] = v;
}

__global__ void copy_k(const float* __restrict__ s, float* __restrict__ d, long n) {
  long i = (long)blockIdx.x * blockDim.x + threadIdx.x;
  if (i < n) d[i] = s[i];
}

__global__ void copy_cols_k(const float* __restrict__ srcM, float* __restrict__ feat,
                            int rows, int cols, int stride, int off) {
  long t = (long)blockIdx.x * blockDim.x + threadIdx.x;
  if (t >= (long)rows * cols) return;
  int r = (int)(t / cols), c = (int)(t - (long)r * cols);
  feat[(long)r * stride + off + c] = srcM[t];
}

// z = eps * exp(0.5*logvar) + mu   (mu == logvar in the reference)
__global__ void z_k(const float* __restrict__ mu, const float* __restrict__ eps,
                    float* __restrict__ z, long n) {
  long i = (long)blockIdx.x * blockDim.x + threadIdx.x;
  if (i < n) z[i] = eps[i] * __expf(0.5f * mu[i]) + mu[i];
}

// cls = log_softmax(m2 @ clsW + clsb) ; m2:[256,32], clsW:[32,2]
__global__ void cls_k(const float* __restrict__ m2, const float* __restrict__ W,
                      const float* __restrict__ b, float* __restrict__ out, int rows) {
  int r = blockIdx.x * blockDim.x + threadIdx.x;
  if (r >= rows) return;
  float l0 = b[0], l1 = b[1];
  for (int k = 0; k < 32; ++k) {
    float v = m2[(long)r * 32 + k];
    l0 += v * W[k * 2 + 0];
    l1 += v * W[k * 2 + 1];
  }
  float mx  = fmaxf(l0, l1);
  float lse = mx + logf(__expf(l0 - mx) + __expf(l1 - mx));
  out[r * 2 + 0] = l0 - lse;
  out[r * 2 + 1] = l1 - lse;
}

// ============================= driver ================================
static inline long cdiv(long a, long b) { return (a + b - 1) / b; }

static void gemm(int act, const float* A, const float* W, const float* b, float* C,
                 int M, int N, int K, hipStream_t s) {
  dim3 grid((unsigned)cdiv(N, 64), (unsigned)(M / 16));
  dim3 blk(128);
  if      (act == 0) gemm_wmma_f16<0><<<grid, blk, 0, s>>>(A, W, b, C, M, N, K);
  else if (act == 1) gemm_wmma_f16<1><<<grid, blk, 0, s>>>(A, W, b, C, M, N, K);
  else if (act == 2) gemm_wmma_f16<2><<<grid, blk, 0, s>>>(A, W, b, C, M, N, K);
  else               gemm_wmma_f16<3><<<grid, blk, 0, s>>>(A, W, b, C, M, N, K);
}

extern "C" void kernel_launch(void* const* d_in, const int* in_sizes, int n_in,
                              void* d_out, int out_size, void* d_ws, size_t ws_size,
                              hipStream_t stream) {
  (void)in_sizes; (void)n_in; (void)out_size; (void)ws_size;

  // ---- inputs (setup_inputs dict order; params flattened in insertion order)
  const float* x    = (const float*)d_in[0];   // [51200,200]
  const int*   ei   = (const int*)  d_in[1];   // [2,E] row-major
  /* batch d_in[2] unused: graph blocks are contiguous */
  const float* ea   = (const float*)d_in[3];   // [E]
  const float* PH1  = (const float*)d_in[4];   // [256,1000]
  const float* PH0  = (const float*)d_in[5];   // [256,1000]
  const float* eps  = (const float*)d_in[6];   // [256,16]
  const float* e0W  = (const float*)d_in[7];   const float* e0b = (const float*)d_in[8];
  const float* e1W  = (const float*)d_in[9];   const float* e1b = (const float*)d_in[10];
  const float* d0W  = (const float*)d_in[11];  const float* d0b = (const float*)d_in[12];
  const float* d1W  = (const float*)d_in[13];  const float* d1b = (const float*)d_in[14];
  const float* ctW1 = (const float*)d_in[15];  const float* ctb1= (const float*)d_in[16];
  const float* ctg1 = (const float*)d_in[17];  const float* ctbb1=(const float*)d_in[18];
  const float* ctg2 = (const float*)d_in[19];  const float* ctbb2=(const float*)d_in[20];
  const float* ctW2 = (const float*)d_in[21];  const float* ctb2= (const float*)d_in[22];
  const float* c1W  = (const float*)d_in[23];  const float* c1b = (const float*)d_in[24];
  const float* c2W  = (const float*)d_in[25];  const float* c2b = (const float*)d_in[26];
  const float* bn0g = (const float*)d_in[27];  const float* bn0b= (const float*)d_in[28];
  const float* fc1W = (const float*)d_in[29];  const float* fc1b= (const float*)d_in[30];
  const float* bn1g = (const float*)d_in[31];  const float* bn1b= (const float*)d_in[32];
  const float* fc2W = (const float*)d_in[33];  const float* fc2b= (const float*)d_in[34];
  const float* clsW = (const float*)d_in[35];  const float* clsb= (const float*)d_in[36];

  const long E = N_EDGES;
  const int* src = ei;
  const int* dst = ei + E;

  // ---- outputs (tuple concatenated flat)
  float* out   = (float*)d_out;
  float* o_cls = out;             // [256,2]
  float* o_mu  = out + 512;       // [256,16]
  float* o_lv  = out + 4608;      // [256,16]
  float* o_z   = out + 8704;      // [256,16]
  float* o_dec = out + 12800;     // [256,1000]
  float* o_bet = out + 268800;    // [256,32]

  // ---- workspace carve-out (floats)
  float* ws = (float*)d_ws;
  size_t o = 0;
  float* hwbuf = ws + o; o += (size_t)N_NODES * 64;   // hw1, later reused as hw2
  float* h1    = ws + o; o += (size_t)N_NODES * 64;
  float* h2    = ws + o; o += (size_t)N_NODES * 16;
  float* dinv  = ws + o; o += N_NODES;
  float* emb   = ws + o; o += 256 * 512;
  float* cbuf  = ws + o; o += 256 * 512;
  float* dec0  = ws + o; o += 256 * 512;
  float* feat  = ws + o; o += 256 * 208;
  float* m1    = ws + o; o += 256 * 256;
  float* m2    = ws + o; o += 256 * 32;
  float* bmean = ws + o; o += 512;
  float* bvar  = ws + o; o += 512;

  const int TPB = 256;

  // ===== GCN normalization: deg (self-loop w=1) -> dinv =====
  fill_k<<<cdiv(N_NODES, TPB), TPB, 0, stream>>>(dinv, 1.0f, N_NODES);
  deg_accum_k<<<cdiv(E, TPB), TPB, 0, stream>>>(dst, ea, dinv, E);
  deg_inv_k<<<cdiv(N_NODES, TPB), TPB, 0, stream>>>(dinv, N_NODES);

  // ===== GCN layer 1: h1 = relu(scatter(x@c1W) + c1b) =====
  gemm(0, x, c1W, nullptr, hwbuf, N_NODES, 64, 200, stream);
  rowbias_init_k<<<cdiv((long)N_NODES * 64, TPB), TPB, 0, stream>>>(h1, c1b, N_NODES, 64);
  gcn_scatter_k<<<cdiv(E * 64, TPB), TPB, 0, stream>>>(hwbuf, src, dst, ea, dinv, h1, E, 64);
  selfloop_k<<<cdiv((long)N_NODES * 64, TPB), TPB, 0, stream>>>(hwbuf, dinv, h1, N_NODES, 64);
  relu_k<<<cdiv((long)N_NODES * 64, TPB), TPB, 0, stream>>>(h1, (long)N_NODES * 64);
  pool_k<<<cdiv(B_GRAPHS * 64, TPB), TPB, 0, stream>>>(h1, feat, 64, 208, 0);    // cols 0..127

  // ===== GCN layer 2: h2 = scatter(h1@c2W) + c2b =====
  gemm(0, h1, c2W, nullptr, hwbuf, N_NODES, 16, 64, stream);
  rowbias_init_k<<<cdiv((long)N_NODES * 16, TPB), TPB, 0, stream>>>(h2, c2b, N_NODES, 16);
  gcn_scatter_k<<<cdiv(E * 16, TPB), TPB, 0, stream>>>(hwbuf, src, dst, ea, dinv, h2, E, 16);
  selfloop_k<<<cdiv((long)N_NODES * 16, TPB), TPB, 0, stream>>>(hwbuf, dinv, h2, N_NODES, 16);
  pool_k<<<cdiv(B_GRAPHS * 16, TPB), TPB, 0, stream>>>(h2, feat, 16, 208, 128);  // cols 128..159

  // ===== PH autoencoder =====
  gemm(2, PH1, e0W, e0b, emb, 256, 512, 1000, stream);           // leaky_relu
  gemm(0, emb, e1W, e1b, o_mu, 256, 16, 512, stream);            // mu
  copy_k<<<cdiv(4096L, TPB), TPB, 0, stream>>>(o_mu, o_lv, 4096); // logvar == mu
  z_k<<<cdiv(4096L, TPB), TPB, 0, stream>>>(o_mu, eps, o_z, 4096);
  copy_cols_k<<<cdiv(4096L, TPB), TPB, 0, stream>>>(o_z, feat, 256, 16, 208, 160);
  gemm(2, o_z, d0W, d0b, dec0, 256, 512, 16, stream);            // leaky_relu
  gemm(3, dec0, d1W, d1b, o_dec, 256, 1000, 512, stream);        // sigmoid

  // ===== CurveTrans: Linear -> BN -> leaky -> BN -> Linear =====
  gemm(0, PH0, ctW1, ctb1, cbuf, 256, 512, 1000, stream);
  bn_stats_k<<<cdiv(512, TPB), TPB, 0, stream>>>(cbuf, bmean, bvar, 256, 512);
  bn_apply_k<2><<<cdiv(256L * 512, TPB), TPB, 0, stream>>>(cbuf, cbuf, bmean, bvar, ctg1, ctbb1, 256, 512);
  bn_stats_k<<<cdiv(512, TPB), TPB, 0, stream>>>(cbuf, bmean, bvar, 256, 512);
  bn_apply_k<0><<<cdiv(256L * 512, TPB), TPB, 0, stream>>>(cbuf, cbuf, bmean, bvar, ctg2, ctbb2, 256, 512);
  gemm(0, cbuf, ctW2, ctb2, o_bet, 256, 32, 512, stream);        // betti
  copy_cols_k<<<cdiv(8192L, TPB), TPB, 0, stream>>>(o_bet, feat, 256, 32, 208, 176);

  // ===== Head MLP =====
  bn_stats_k<<<cdiv(208, TPB), TPB, 0, stream>>>(feat, bmean, bvar, 256, 208);
  bn_apply_k<0><<<cdiv(256L * 208, TPB), TPB, 0, stream>>>(feat, feat, bmean, bvar, bn0g, bn0b, 256, 208);
  gemm(1, feat, fc1W, fc1b, m1, 256, 256, 208, stream);          // relu
  bn_stats_k<<<cdiv(256, TPB), TPB, 0, stream>>>(m1, bmean, bvar, 256, 256);
  bn_apply_k<0><<<cdiv(256L * 256, TPB), TPB, 0, stream>>>(m1, m1, bmean, bvar, bn1g, bn1b, 256, 256);
  gemm(1, m1, fc2W, fc2b, m2, 256, 32, 256, stream);             // relu
  cls_k<<<cdiv(256, TPB), TPB, 0, stream>>>(m2, clsW, clsb, o_cls, 256);
}
